// RelativeLearnableAttention_12292196401635
// MI455X (gfx1250) — matrix operand order repaired and checked by reading, not compiled, not run on hardware
//
#include <hip/hip_runtime.h>
#include <float.h>

#define B_ 4
#define N_ 2048
#define DIM_ 256
#define H_ 8
#define D_ 32
#define SSTRIDE 2052   // 2048 + 4 : multiple of 4 (float4), 4*row bank skew (all 16 rows distinct banks)

typedef __attribute__((ext_vector_type(16))) __bf16 v16bf;
typedef __attribute__((ext_vector_type(8)))  float  v8f;
typedef __attribute__((ext_vector_type(4)))  unsigned int v4u;
typedef __attribute__((ext_vector_type(4)))  float  v4f;

__device__ __forceinline__ unsigned short f2bf(float f) {
  unsigned u = __builtin_bit_cast(unsigned, f);
  unsigned r = u + 0x7FFFu + ((u >> 16) & 1u);   // round-to-nearest-even
  return (unsigned short)(r >> 16);
}
__device__ __forceinline__ float bf2f(unsigned short h) {
  return __builtin_bit_cast(float, (unsigned)h << 16);
}

union FragBF {                 // one WMMA 16x16x32 bf16 operand: 16 halfs / lane
  v16bf v;
  unsigned short u[16];
  v4u q[2];                    // two 16B chunks
};

__device__ __forceinline__ v8f wmma_bf16(v16bf a, v16bf b, v8f c) {
  return __builtin_amdgcn_wmma_f32_16x16x32_bf16(false, a, false, b, (short)0, c,
                                                 false, false);
}

// ---------------- fp32 -> bf16 conversion kernels ----------------
__global__ void cvt_rm_kernel(const float* __restrict__ in,
                              unsigned short* __restrict__ out, int n) {
  int i = blockIdx.x * blockDim.x + threadIdx.x;
  if (i < n) out[i] = f2bf(in[i]);
}

// out[c][r] = in[r][c]  (store B-operand weights transposed)
__global__ void cvt_tr_kernel(const float* __restrict__ in,
                              unsigned short* __restrict__ out, int R, int C) {
  int i = blockIdx.x * blockDim.x + threadIdx.x;
  if (i < R * C) {
    int r = i / C, c = i % C;
    out[(size_t)c * R + r] = f2bf(in[i]);
  }
}

// ---------------- QKV GEMM: w[b] (N x 256) @ Wqkv (256 x 768) ----------------
// one wave per 16x16 output tile; scatter into [b,h,n,d] bf16 head layout
__global__ void qkv_gemm_kernel(const unsigned short* __restrict__ wbf,
                                const unsigned short* __restrict__ WqkvT,
                                unsigned short* __restrict__ qh,
                                unsigned short* __restrict__ kh,
                                unsigned short* __restrict__ vh) {
  const int lane = threadIdx.x;
  const int b  = blockIdx.z;
  const int i0 = blockIdx.y * 16;
  const int c0 = blockIdx.x * 16;
  const int nn   = lane & 15;
  const int hiA  = (lane & 16) ? 16 : 0;  // A k-offset per half-wave
  const int hiB  = (lane & 16) ? 16 : 0;  // B k-offset per half-wave

  const unsigned short* arow = wbf + ((size_t)b * N_ + (i0 + nn)) * DIM_;
  const unsigned short* brow = WqkvT + (size_t)(c0 + nn) * DIM_;

  v8f acc = {};
  #pragma unroll
  for (int k0 = 0; k0 < DIM_; k0 += 32) {
    FragBF a, bb;
    a.q[0]  = *(const v4u*)(arow + k0 + hiA);
    a.q[1]  = *(const v4u*)(arow + k0 + hiA + 8);
    bb.q[0] = *(const v4u*)(brow + k0 + hiB);
    bb.q[1] = *(const v4u*)(brow + k0 + hiB + 8);
    acc = wmma_bf16(a.v, bb.v, acc);
  }
  const int mbase = (lane & 16) ? 8 : 0;
  #pragma unroll
  for (int r = 0; r < 8; ++r) {
    int i  = i0 + mbase + r;
    int c  = c0 + nn;
    int which = c >> 8;         // 0=q 1=k 2=v (tile never crosses a 256 boundary)
    int cc = c & 255;
    int hh = cc >> 5, dd = cc & 31;
    size_t idx = (((size_t)b * H_ + hh) * N_ + i) * D_ + dd;
    unsigned short val = f2bf(acc[r]);
    if (which == 0)      qh[idx] = val;
    else if (which == 1) kh[idx] = val;
    else                 vh[idx] = val;
  }
}

// ---------------- rp GEMM: r (2048 x 256) @ Wr (256 x 256), flat row-major ----
__global__ void rp_gemm_kernel(const unsigned short* __restrict__ rbf,
                               const unsigned short* __restrict__ WrT,
                               unsigned short* __restrict__ rp) {
  const int lane = threadIdx.x;
  const int i0 = blockIdx.y * 16;
  const int c0 = blockIdx.x * 16;
  const int nn  = lane & 15;
  const int hiA = (lane & 16) ? 16 : 0;
  const int hiB = (lane & 16) ? 16 : 0;

  const unsigned short* arow = rbf + (size_t)(i0 + nn) * DIM_;
  const unsigned short* brow = WrT + (size_t)(c0 + nn) * DIM_;

  v8f acc = {};
  #pragma unroll
  for (int k0 = 0; k0 < DIM_; k0 += 32) {
    FragBF a, bb;
    a.q[0]  = *(const v4u*)(arow + k0 + hiA);
    a.q[1]  = *(const v4u*)(arow + k0 + hiA + 8);
    bb.q[0] = *(const v4u*)(brow + k0 + hiB);
    bb.q[1] = *(const v4u*)(brow + k0 + hiB + 8);
    acc = wmma_bf16(a.v, bb.v, acc);
  }
  const int mbase = (lane & 16) ? 8 : 0;
  #pragma unroll
  for (int r = 0; r < 8; ++r)
    rp[(size_t)(i0 + mbase + r) * DIM_ + c0 + nn] = f2bf(acc[r]);
}

// ---------------- attention: per (b,h,16-row strip) ----------------
__global__ __launch_bounds__(128) void attn_kernel(
    const unsigned short* __restrict__ qh,
    const unsigned short* __restrict__ kh,
    const unsigned short* __restrict__ vh,
    const unsigned short* __restrict__ rp,
    const int*  __restrict__ mask,             // [B, N-1] (integer: nonzero = keep)
    const float* __restrict__ rwb,             // [H,1,D]
    const float* __restrict__ rrb,             // [H,1,D]
    float* __restrict__ attn_out,              // [B,H,N,N]
    unsigned short* __restrict__ ctx) {        // [B,N,DIM] bf16
  __shared__ __attribute__((aligned(16))) float sS[16 * SSTRIDE];   // ~128 KB (CDNA5 320KB LDS)
  __shared__ unsigned short sQw[16 * 32];
  __shared__ unsigned short sQr[16 * 32];
  __shared__ float sRed[16 * 8];
  __shared__ float sRow[16];
  __shared__ unsigned char sMrow[16];
  __shared__ unsigned char sMcol[N_];
  __shared__ __attribute__((aligned(16))) float sPart[4 * 16 * D_];

  const int tid  = threadIdx.x;
  const int lane = tid & 31;
  const int wave = tid >> 5;
  const int bid  = blockIdx.x;
  const int it = bid & 127;
  const int h  = (bid >> 7) & 7;
  const int b  = bid >> 10;
  const int i0 = it * 16;
  const float scale = 0.0625f;               // 256^-0.5
  const size_t bh = (size_t)b * H_ + h;

  // stage q tile with both biases (bf16) + masks
  for (int idx = tid; idx < 16 * 32; idx += 128) {
    int r = idx >> 5, d = idx & 31;
    float qv = bf2f(qh[(bh * N_ + (i0 + r)) * D_ + d]);
    sQw[idx] = f2bf(qv + rwb[h * D_ + d]);
    sQr[idx] = f2bf(qv + rrb[h * D_ + d]);
  }
  for (int j = tid; j < N_; j += 128)
    sMcol[j] = (j == 0) ? (unsigned char)1
                        : (unsigned char)(mask[(size_t)b * (N_ - 1) + (j - 1)] != 0);
  if (tid < 16) {
    int i = i0 + tid;
    sMrow[tid] = (i == 0) ? (unsigned char)1
                          : (unsigned char)(mask[(size_t)b * (N_ - 1) + (i - 1)] != 0);
  }
  __syncthreads();

  // A fragments (Q+rw_bias, Q+rr_bias) — identical per wave, reused for all j-tiles
  FragBF aQw, aQr;
  {
    const int m = lane & 15;
    const int koff = (lane & 16) ? 16 : 0;
    #pragma unroll
    for (int e = 0; e < 16; ++e) {
      int kk = e + koff;
      aQw.u[e] = sQw[m * 32 + kk];
      aQr.u[e] = sQr[m * 32 + kk];
    }
  }

  // scores: S = (Q+rw)·K^T + (Q+rr)·rp^T ; fused into one accumulator chain
  {
    const int nn   = lane & 15;
    const int koff = (lane & 16) ? 16 : 0;
    const int mb   = (lane & 16) ? 8 : 0;
    for (int jt = wave; jt < N_ / 16; jt += 4) {
      int j0 = jt * 16;
      const unsigned short* kpp = kh + (bh * N_ + (j0 + nn)) * D_ + koff;
      const unsigned short* rpp = rp + ((size_t)h * N_ + (j0 + nn)) * D_ + koff;
      // prefetch next j-tile for this wave (jt+4 -> +64 rows): global_prefetch_b8
      if (j0 + 64 < N_) {
        __builtin_prefetch(kpp + 64 * D_, 0, 0);
        __builtin_prefetch(rpp + 64 * D_, 0, 0);
      }
      FragBF bK, bR;
      bK.q[0] = *(const v4u*)(kpp);  bK.q[1] = *(const v4u*)(kpp + 8);
      bR.q[0] = *(const v4u*)(rpp);  bR.q[1] = *(const v4u*)(rpp + 8);
      v8f z = {};
      v8f acc = wmma_bf16(aQr.v, bR.v, z);
      acc = wmma_bf16(aQw.v, bK.v, acc);
      #pragma unroll
      for (int r = 0; r < 8; ++r)
        sS[(mb + r) * SSTRIDE + j0 + nn] = acc[r];
    }
  }
  __syncthreads();

  // masked softmax over rows (8 threads x 256 cols per row)
  const int row = tid & 15;
  const int seg = tid >> 4;
  const int jbase = seg * 256;
  const int mrow = sMrow[row];

  float mx = -FLT_MAX;
  for (int jj = 0; jj < 256; ++jj) {
    int j = jbase + jj;
    float v = sS[row * SSTRIDE + j] * scale;
    v = (mrow && sMcol[j]) ? v : -FLT_MAX;
    sS[row * SSTRIDE + j] = v;
    mx = fmaxf(mx, v);
  }
  sRed[row * 8 + seg] = mx;
  __syncthreads();
  if (tid < 16) {
    float m2 = -FLT_MAX;
    for (int s2 = 0; s2 < 8; ++s2) m2 = fmaxf(m2, sRed[tid * 8 + s2]);
    sRow[tid] = m2;
  }
  __syncthreads();
  const float rmax = sRow[row];
  float sum = 0.f;
  for (int jj = 0; jj < 256; ++jj) {
    int j = jbase + jj;
    float e = __expf(sS[row * SSTRIDE + j] - rmax);
    sS[row * SSTRIDE + j] = e;
    sum += e;
  }
  sRed[row * 8 + seg] = sum;
  __syncthreads();
  if (tid < 16) {
    float s3 = 0.f;
    for (int s2 = 0; s2 < 8; ++s2) s3 += sRed[tid * 8 + s2];
    sRow[tid] = 1.0f / s3;
  }
  __syncthreads();
  const float rinv = sRow[row];
  for (int jj = 0; jj < 256; ++jj)
    sS[row * SSTRIDE + jbase + jj] *= rinv;
  __syncthreads();

  // coalesced float4 streaming write of attn (the 537 MB that dominates runtime)
  for (int r = 0; r < 16; ++r) {
    const size_t base = (bh * N_ + (i0 + r)) * (size_t)N_;
    for (int j = tid * 4; j < N_; j += 128 * 4)
      *(v4f*)(attn_out + base + j) = *(const v4f*)(&sS[r * SSTRIDE + j]);
  }

  // PV: out[16x32] = P[16x2048] @ V[2048x32], K split across 4 waves
  {
    v8f acc0 = {}, acc1 = {};
    const int nn    = lane & 15;
    const int koffB = (lane & 16) ? 16 : 0;
    const int koffA = (lane & 16) ? 16 : 0;
    const int m     = lane & 15;
    for (int jt = wave; jt < N_ / 32; jt += 4) {
      int j0 = jt * 32;
      // prefetch this wave's next V chunk (jt+4 -> +128 rows)
      if (j0 + 128 < N_)
        __builtin_prefetch(vh + (bh * N_ + (j0 + 128 + koffB)) * D_, 0, 0);
      FragBF aP, bv0, bv1;
      #pragma unroll
      for (int e = 0; e < 16; ++e) {
        int kk = e + koffA;
        aP.u[e] = f2bf(sS[m * SSTRIDE + j0 + kk]);
      }
      #pragma unroll
      for (int e = 0; e < 16; ++e) {
        const unsigned short* vrow = vh + (bh * N_ + (j0 + koffB + e)) * D_;
        bv0.u[e] = vrow[nn];
        bv1.u[e] = vrow[16 + nn];
      }
      acc0 = wmma_bf16(aP.v, bv0.v, acc0);
      acc1 = wmma_bf16(aP.v, bv1.v, acc1);
    }
    float* sp = sPart + wave * (16 * D_);
    const int mb = (lane & 16) ? 8 : 0;
    #pragma unroll
    for (int r = 0; r < 8; ++r) {
      sp[(mb + r) * D_ + nn]      = acc0[r];
      sp[(mb + r) * D_ + 16 + nn] = acc1[r];
    }
  }
  __syncthreads();
  for (int idx = tid; idx < 16 * D_; idx += 128) {
    float s4 = sPart[idx] + sPart[512 + idx] + sPart[1024 + idx] + sPart[1536 + idx];
    int r = idx >> 5, d = idx & 31;
    ctx[((size_t)b * N_ + (i0 + r)) * DIM_ + h * D_ + d] = f2bf(s4);
  }
}

// ---------------- out GEMM: ctx (B*N x 256) @ Wout (256 x 256) -> fp32 -------
__global__ void out_gemm_kernel(const unsigned short* __restrict__ ctx,
                                const unsigned short* __restrict__ WoutT,
                                float* __restrict__ out) {
  const int lane = threadIdx.x;
  const int i0 = blockIdx.y * 16;    // rows over B*N
  const int c0 = blockIdx.x * 16;
  const int nn  = lane & 15;
  const int hiA = (lane & 16) ? 16 : 0;
  const int hiB = (lane & 16) ? 16 : 0;

  const unsigned short* arow = ctx + (size_t)(i0 + nn) * DIM_;
  const unsigned short* brow = WoutT + (size_t)(c0 + nn) * DIM_;

  v8f acc = {};
  #pragma unroll
  for (int k0 = 0; k0 < DIM_; k0 += 32) {
    FragBF a, bb;
    a.q[0]  = *(const v4u*)(arow + k0 + hiA);
    a.q[1]  = *(const v4u*)(arow + k0 + hiA + 8);
    bb.q[0] = *(const v4u*)(brow + k0 + hiB);
    bb.q[1] = *(const v4u*)(brow + k0 + hiB + 8);
    acc = wmma_bf16(a.v, bb.v, acc);
  }
  const int mbase = (lane & 16) ? 8 : 0;
  #pragma unroll
  for (int r = 0; r < 8; ++r)
    out[(size_t)(i0 + mbase + r) * DIM_ + c0 + nn] = acc[r];
}

// ---------------- launch ----------------
extern "C" void kernel_launch(void* const* d_in, const int* in_sizes, int n_in,
                              void* d_out, int out_size, void* d_ws, size_t ws_size,
                              hipStream_t stream) {
  const float* w    = (const float*)d_in[0];
  const float* r    = (const float*)d_in[1];
  const int*   mask = (const int*)d_in[2];
  const float* Wqkv = (const float*)d_in[3];
  const float* Wr   = (const float*)d_in[4];
  const float* Wout = (const float*)d_in[5];
  const float* rwb  = (const float*)d_in[6];
  const float* rrb  = (const float*)d_in[7];

  unsigned short* p = (unsigned short*)d_ws;
  unsigned short* wbf   = p; p += (size_t)B_ * N_ * DIM_;
  unsigned short* rbf   = p; p += (size_t)N_ * DIM_;
  unsigned short* WqkvT = p; p += (size_t)3 * DIM_ * DIM_;
  unsigned short* WrT   = p; p += (size_t)DIM_ * DIM_;
  unsigned short* WoutT = p; p += (size_t)DIM_ * DIM_;
  unsigned short* qh    = p; p += (size_t)B_ * H_ * N_ * D_;
  unsigned short* kh    = p; p += (size_t)B_ * H_ * N_ * D_;
  unsigned short* vh    = p; p += (size_t)B_ * H_ * N_ * D_;
  unsigned short* rph   = p; p += (size_t)N_ * DIM_;
  unsigned short* ctx   = p; p += (size_t)B_ * N_ * DIM_;

  float* out  = (float*)d_out;
  float* attn = out + (size_t)B_ * N_ * DIM_;

  const int nW = B_ * N_ * DIM_;
  const int nR = N_ * DIM_;
  cvt_rm_kernel<<<(nW + 255) / 256, 256, 0, stream>>>(w, wbf, nW);
  cvt_rm_kernel<<<(nR + 255) / 256, 256, 0, stream>>>(r, rbf, nR);
  cvt_tr_kernel<<<(3 * DIM_ * DIM_ + 255) / 256, 256, 0, stream>>>(Wqkv, WqkvT, DIM_, 3 * DIM_);
  cvt_tr_kernel<<<(DIM_ * DIM_ + 255) / 256, 256, 0, stream>>>(Wr, WrT, DIM_, DIM_);
  cvt_tr_kernel<<<(DIM_ * DIM_ + 255) / 256, 256, 0, stream>>>(Wout, WoutT, DIM_, DIM_);

  qkv_gemm_kernel<<<dim3(3 * DIM_ / 16, N_ / 16, B_), 32, 0, stream>>>(wbf, WqkvT, qh, kh, vh);
  rp_gemm_kernel<<<dim3(DIM_ / 16, N_ / 16), 32, 0, stream>>>(rbf, WrT, rph);

  attn_kernel<<<dim3(B_ * H_ * (N_ / 16)), 128, 0, stream>>>(qh, kh, vh, rph, mask,
                                                             rwb, rrb, attn, ctx);

  out_gemm_kernel<<<dim3(DIM_ / 16, (B_ * N_) / 16), 32, 0, stream>>>(ctx, WoutT, out);
}
